// VRNN_43593918054783
// MI455X (gfx1250) — matrix operand order repaired and checked by reading, not compiled, not run on hardware
//
#include <hip/hip_runtime.h>
#include <hip/hip_bf16.h>
#include <math.h>

#define T_  256
#define B_  512
#define XD_ 64
#define HD_ 512
#define ZD_ 32
#define EPSC 1e-5f

typedef __attribute__((ext_vector_type(16))) __bf16          v16bf;
typedef __attribute__((ext_vector_type(8)))  float           v8f;
typedef __attribute__((ext_vector_type(8)))  unsigned short  ushort8;
typedef __attribute__((ext_vector_type(16))) unsigned short  ushort16;
typedef __attribute__((ext_vector_type(4)))  unsigned short  ushort4v;
typedef __attribute__((ext_vector_type(4)))  unsigned int    uint4v;
typedef __attribute__((ext_vector_type(8)))  int             int8v;
typedef __attribute__((ext_vector_type(4)))  int             int4v;

typedef unsigned short u16;

__device__ __forceinline__ unsigned short f2bf(float f) {
    __bf16 b = (__bf16)f;                       // hardware RNE convert
    return __builtin_bit_cast(unsigned short, b);
}
__device__ __forceinline__ float bf2f(unsigned short u) {
    return (float)__builtin_bit_cast(__bf16, u);
}

// ---------------------------------------------------------------------------
// TDM: one 2D-tile DMA (global bf16, row stride ld_elems) -> LDS byte offset.
// Tile: rows x 32 elements, 2-byte data. OOB rows (W tiles, N=32/64) read zero.
// Descriptor packing per CDNA5 ISA 8.3/8.4 (D# group0/group1), groups 2/3 = 0.
// This toolchain uses the 6-arg builtin (extra int32x8 group before cpol).
// ---------------------------------------------------------------------------
__device__ __forceinline__ void tdm_load_tile(unsigned lds_addr, const u16* gptr,
                                              int rows, int ld_elems) {
    unsigned long long ga = (unsigned long long)(const void*)gptr;
    uint4v g0;
    g0[0] = 1u;                                            // count=1 (valid, user)
    g0[1] = lds_addr;                                      // lds_addr bytes [63:32]
    g0[2] = (unsigned)(ga & 0xFFFFFFFFu);                  // global_addr[31:0]
    g0[3] = (unsigned)((ga >> 32) & 0x1FFFFFFu)            // global_addr[56:32]
          | (2u << 30);                                    // type=2 ("image")
    int8v g1;
    g1[0] = (1 << 16);                                     // wg_mask=0, data_size=1 (2B)
    g1[1] = (32 << 16);                                    // tensor_dim0[15:0]=32
    g1[2] = (rows << 16);                                  // dim0[31:16]=0 | tensor_dim1[15:0]=rows
    g1[3] = (32 << 16);                                    // dim1[31:16]=0 | tile_dim0=32
    g1[4] = rows & 0xFFFF;                                 // tile_dim1=rows, tile_dim2=0
    g1[5] = ld_elems;                                      // tensor_dim0_stride[31:0]
    g1[6] = 0;                                             // stride hi | dim1_stride lo
    g1[7] = 0;                                             // dim1_stride hi
    int4v z4 = {0, 0, 0, 0};
    int8v z8 = {0, 0, 0, 0, 0, 0, 0, 0};
    __builtin_amdgcn_tensor_load_to_lds(g0, g1, z4, z4, z8, 0);
}

// ---------------------------------------------------------------------------
// Fused GEMM on bf16 operands: C[M,N] = act( A[M,K] @ W[N,K]^T + bias (+ add) )
// Block tile 128x64, BK=32, 256 threads = 8 waves; wave = 32x32 tile (2x2 wmma).
// Tiles moved by the Tensor Data Mover into double-buffered dynamic LDS:
//   A: bytes [0,8K) / [8K,16K) ; W: [16K,20K) / [20K,24K)
// ACT: 0=none 1=relu 2=softplus ; OUTBF selects f32 or bf16 C.
// Requires M%128==0, K%32==0; N handled via TDM zero-fill + store guard.
// ---------------------------------------------------------------------------
template<int ACT, bool ADD, bool OUTBF>
__global__ __launch_bounds__(256)
void gemm_bf16_wmma(const u16* __restrict__ A, int lda,
                    const u16* __restrict__ W, int ldw,
                    const float* __restrict__ bias,
                    const void* addsrc,
                    void* Cv, int ldc,
                    int N, int K)
{
    extern __shared__ __align__(16) unsigned short smem[];

    const int tid  = threadIdx.x;
    const int lane = tid & 31;
    const int wave = tid >> 5;
    const int wm   = wave & 3;     // wave row tile (32 rows)
    const int wn   = wave >> 2;    // wave col tile (32 cols)
    const int row0 = blockIdx.x * 128;
    const int col0 = blockIdx.y * 64;
    const int wrows = (N - col0 < 64) ? (N - col0) : 64;

    auto issue = [&](int k0, int buf) {
        if (wave == 0) {
            tdm_load_tile(buf ? 8192u : 0u,      A + row0 * lda + k0, 128,   lda);
            tdm_load_tile(buf ? 20480u : 16384u, W + col0 * ldw + k0, wrows, ldw);
        }
    };

    v8f acc[2][2] = {};

    issue(0, 0);
    if (wave == 0) __builtin_amdgcn_s_wait_tensorcnt(0);
    __syncthreads();

    const int nk  = K >> 5;
    const int mr  = lane & 15;
    const int kba = (lane >> 4) << 3;   // A frag K split per ISA layout
    const int kbb = (lane >> 4) << 4;   // unused: B frag is loaded with kba, same K split as A

    for (int ki = 0; ki < nk; ++ki) {
        const int cur = ki & 1;
        if (ki + 1 < nk) issue((ki + 1) << 5, cur ^ 1);

        const unsigned short* Ab = smem + (cur ? 4096 : 0);         // element idx
        const unsigned short* Wb = smem + 8192 + (cur ? 2048 : 0);

        v16bf af[2], bfm[2];
#pragma unroll
        for (int s = 0; s < 2; ++s) {
            const unsigned short* ap = &Ab[(wm * 32 + s * 16 + mr) * 32];
            ushort8 alo = *(const ushort8*)(ap + kba);
            ushort8 ahi = *(const ushort8*)(ap + kba + 16);
            const unsigned short* bp = &Wb[(wn * 32 + s * 16 + mr) * 32];
            ushort8 blo = *(const ushort8*)(bp + kba);
            ushort8 bhi = *(const ushort8*)(bp + kba + 16);
            ushort16 av, bv;
#pragma unroll
            for (int i = 0; i < 8; ++i) {
                av[i] = alo[i]; av[i + 8] = ahi[i];
                bv[i] = blo[i]; bv[i + 8] = bhi[i];
            }
            af[s]  = __builtin_bit_cast(v16bf, av);
            bfm[s] = __builtin_bit_cast(v16bf, bv);
        }
#pragma unroll
        for (int i = 0; i < 2; ++i)
#pragma unroll
            for (int j = 0; j < 2; ++j)
                acc[i][j] = __builtin_amdgcn_wmma_f32_16x16x32_bf16(
                        false, af[i], false, bfm[j], (short)0, acc[i][j], false, false);

        if (wave == 0) __builtin_amdgcn_s_wait_tensorcnt(0);
        __syncthreads();
    }

    // epilogue: VGPR v -> M = v (+8 for lanes>=16), N = lane&15
#pragma unroll
    for (int i = 0; i < 2; ++i) {
        const int rbase = row0 + wm * 32 + i * 16 + ((lane >> 4) << 3);
#pragma unroll
        for (int j = 0; j < 2; ++j) {
            const int cn = col0 + wn * 32 + j * 16 + (lane & 15);
            if (cn < N) {
                const float bv = bias ? bias[cn] : 0.0f;
#pragma unroll
                for (int v = 0; v < 8; ++v) {
                    const int idx = (rbase + v) * ldc + cn;
                    float xv = acc[i][j][v] + bv;
                    if (ADD) {
                        if (OUTBF) xv += bf2f(((const unsigned short*)addsrc)[idx]);
                        else       xv += ((const float*)addsrc)[idx];
                    }
                    if (ACT == 1) xv = fmaxf(xv, 0.0f);
                    if (ACT == 2) xv = fmaxf(xv, 0.0f) + log1pf(expf(-fabsf(xv)));
                    if (OUTBF) ((unsigned short*)Cv)[idx] = f2bf(xv);
                    else       ((float*)Cv)[idx] = xv;
                }
            }
        }
    }
}

// ---------------------------------------------------------------------------
// Elementwise / conversion / reduction kernels
// ---------------------------------------------------------------------------
__global__ void k_f2bf4(const float* __restrict__ src, unsigned short* __restrict__ dst, int n4) {
    int i = blockIdx.x * blockDim.x + threadIdx.x;
    if (i >= n4) return;
    float4 f = *(const float4*)(src + i * 4);
    ushort4v u;
    u[0] = f2bf(f.x); u[1] = f2bf(f.y); u[2] = f2bf(f.z); u[3] = f2bf(f.w);
    *(ushort4v*)(dst + i * 4) = u;
}

__global__ void k_zero_f(float* p, int n) {
    int i = blockIdx.x * blockDim.x + threadIdx.x;
    if (i < n) p[i] = 0.0f;
}
__global__ void k_zero_h(unsigned short* p, int n) {
    int i = blockIdx.x * blockDim.x + threadIdx.x;
    if (i < n) p[i] = 0;
}

__global__ void k_make_z(const float* __restrict__ eps_t,
                         const float* __restrict__ mean,
                         const float* __restrict__ stdv,
                         unsigned short* __restrict__ z, int n) {
    int i = blockIdx.x * blockDim.x + threadIdx.x;
    if (i >= n) return;
    float v = eps_t[i] * stdv[i] + mean[i];
    if ((i & (ZD_ - 1)) < 4) v *= 10.0f;     // _modify_z dims 0..3 * 10
    z[i] = f2bf(v);
}

__global__ void k_gru(const float* __restrict__ gx, const float* __restrict__ gh,
                      const unsigned short* __restrict__ h,
                      unsigned short* __restrict__ hnew, int n) {
    int i = blockIdx.x * blockDim.x + threadIdx.x;
    if (i >= n) return;
    int b = i / HD_, c = i % HD_;
    const float* gxr = gx + b * 3 * HD_;
    const float* ghr = gh + b * 3 * HD_;
    float r  = 1.0f / (1.0f + expf(-(gxr[c] + ghr[c])));
    float zg = 1.0f / (1.0f + expf(-(gxr[HD_ + c] + ghr[HD_ + c])));
    float nn = tanhf(gxr[2 * HD_ + c] + r * ghr[2 * HD_ + c]);
    hnew[i] = f2bf((1.0f - zg) * nn + zg * bf2f(h[i]));
}

__device__ __forceinline__ float block_sum(float v, float* red) {
    red[threadIdx.x] = v; __syncthreads();
    for (int s = 128; s > 0; s >>= 1) {
        if (threadIdx.x < (unsigned)s) red[threadIdx.x] += red[threadIdx.x + s];
        __syncthreads();
    }
    return red[0];
}

__global__ void k_kld(const float* __restrict__ m1, const float* __restrict__ s1p,
                      const float* __restrict__ m2, const float* __restrict__ s2p,
                      float* out, int n) {
    __shared__ float red[256];
    int i = blockIdx.x * blockDim.x + threadIdx.x;
    float v = 0.0f;
    if (i < n) {
        float s1 = fmaxf(s1p[i], 1e-9f);
        float s2 = fmaxf(s2p[i], 1e-9f);
        float d  = m1[i] - m2[i];
        v = 0.5f * (2.0f * (logf(s2) - logf(s1)) + (s1 * s1 + d * d) / (s2 * s2) - 1.0f);
    }
    float s = block_sum(v, red);
    if (threadIdx.x == 0) atomicAdd(&out[1], s);
}

__global__ void k_nll_rec(const float* __restrict__ x_t, const float* __restrict__ dmean,
                          const float* __restrict__ dstd, float* out, int n) {
    __shared__ float red[256];
    int i = blockIdx.x * blockDim.x + threadIdx.x;
    float nll = 0.0f, rec = 0.0f;
    if (i < n) {
        float xv = x_t[i], m = dmean[i], s = dstd[i];
        float d  = xv - m;
        nll = logf(s + EPSC) + 0.91893853320467274f + d * d / (2.0f * s * s);
        rec = d * d;
    }
    float s1 = block_sum(nll, red); __syncthreads();
    float s2 = block_sum(rec, red);
    if (threadIdx.x == 0) { atomicAdd(&out[2], s1); atomicAdd(&out[0], s2); }
}

// ---------------------------------------------------------------------------
// Host orchestration
// ---------------------------------------------------------------------------
static inline void gemm(int act, bool add, bool outbf, hipStream_t st,
                        const u16* A, int lda, const u16* W, int ldw,
                        const float* bias, const void* add_p,
                        void* C, int ldc, int M, int N, int K) {
    dim3 g(M / 128, (N + 63) / 64), b(256);
    const size_t shmem = 24576;   // 2x(8K A) + 2x(4K W)
#define LG(ACT, ADD, OBF) hipLaunchKernelGGL((gemm_bf16_wmma<ACT, ADD, OBF>), g, b, shmem, st, \
                                             A, lda, W, ldw, bias, add_p, C, ldc, N, K)
    if (outbf) {
        if (add) { if (act == 0) LG(0, true,  true); else if (act == 1) LG(1, true,  true); else LG(2, true,  true); }
        else     { if (act == 0) LG(0, false, true); else if (act == 1) LG(1, false, true); else LG(2, false, true); }
    } else {
        if (add) { if (act == 0) LG(0, true,  false); else if (act == 1) LG(1, true,  false); else LG(2, true,  false); }
        else     { if (act == 0) LG(0, false, false); else if (act == 1) LG(1, false, false); else LG(2, false, false); }
    }
#undef LG
}

static inline void cvt(hipStream_t st, const float* src, u16* dst, int n) {
    int n4 = n / 4;
    hipLaunchKernelGGL(k_f2bf4, dim3((n4 + 255) / 256), dim3(256), 0, st, src, dst, n4);
}

extern "C" void kernel_launch(void* const* d_in, const int* in_sizes, int n_in,
                              void* d_out, int out_size, void* d_ws, size_t ws_size,
                              hipStream_t stream) {
    const float* x   = (const float*)d_in[0];
    const float* eps = (const float*)d_in[1];
    const float* biasv[34];
    for (int i = 0; i < 34; ++i) biasv[i] = (const float*)d_in[i];

    float* out = (float*)d_out;

    // ---- workspace carve-up (256B aligned chunks) ----
    char* base = (char*)d_ws;
    size_t off = 0;
    auto allocB = [&](size_t bytes) { void* p = base + off; off += (bytes + 255) & ~(size_t)255; return p; };
    auto allocF = [&](size_t n) { return (float*)allocB(n * 4); };
    auto allocH = [&](size_t n) { return (u16*)allocB(n * 2); };

    // bf16 weight copies
    u16* W_phi_x1 = allocH((size_t)HD_ * XD_);
    u16* W_phi_x2 = allocH((size_t)HD_ * HD_);
    u16* W_phi_z  = allocH((size_t)HD_ * ZD_);
    u16* W_enc1   = allocH((size_t)HD_ * 2 * HD_);
    u16* W_enc2   = allocH((size_t)HD_ * HD_);
    u16* W_encm   = allocH((size_t)ZD_ * HD_);
    u16* W_encs   = allocH((size_t)ZD_ * HD_);
    u16* W_pri    = allocH((size_t)HD_ * HD_);
    u16* W_prim   = allocH((size_t)ZD_ * HD_);
    u16* W_pris   = allocH((size_t)ZD_ * HD_);
    u16* W_dec1   = allocH((size_t)HD_ * 2 * HD_);
    u16* W_dec2   = allocH((size_t)HD_ * HD_);
    u16* W_decs   = allocH((size_t)XD_ * HD_);
    u16* W_dm1    = allocH((size_t)XD_ * HD_);
    u16* W_dm2    = allocH((size_t)XD_ * XD_);
    u16* W_gih    = allocH((size_t)3 * HD_ * 2 * HD_);
    u16* W_ghh    = allocH((size_t)3 * HD_ * HD_);
    // bf16 activations
    u16* x_bf     = allocH((size_t)T_ * B_ * XD_);
    u16* tmp1     = allocH((size_t)B_ * HD_);
    u16* phi_x    = allocH((size_t)B_ * HD_);
    u16* enc_h1   = allocH((size_t)B_ * HD_);
    u16* enc_h    = allocH((size_t)B_ * HD_);
    u16* pri_h    = allocH((size_t)B_ * HD_);
    u16* dec_h1   = allocH((size_t)B_ * HD_);
    u16* dec_h    = allocH((size_t)B_ * HD_);
    u16* phi_z    = allocH((size_t)B_ * HD_);
    u16* h0       = allocH((size_t)B_ * HD_);
    u16* h1       = allocH((size_t)B_ * HD_);
    u16* zbuf     = allocH((size_t)B_ * ZD_);
    u16* dm1      = allocH((size_t)B_ * XD_);
    // f32 buffers
    float* gx     = allocF((size_t)B_ * 3 * HD_);
    float* gh     = allocF((size_t)B_ * 3 * HD_);
    float* e_mean = allocF((size_t)B_ * ZD_);
    float* e_std  = allocF((size_t)B_ * ZD_);
    float* p_mean = allocF((size_t)B_ * ZD_);
    float* p_std  = allocF((size_t)B_ * ZD_);
    float* dstd   = allocF((size_t)B_ * XD_);

    // ---- one-time per call: convert weights + x to bf16, zero h/losses ----
    cvt(stream, (const float*)d_in[2],  W_phi_x1, HD_ * XD_);
    cvt(stream, (const float*)d_in[4],  W_phi_x2, HD_ * HD_);
    cvt(stream, (const float*)d_in[6],  W_phi_z,  HD_ * ZD_);
    cvt(stream, (const float*)d_in[8],  W_enc1,   HD_ * 2 * HD_);
    cvt(stream, (const float*)d_in[10], W_enc2,   HD_ * HD_);
    cvt(stream, (const float*)d_in[12], W_encm,   ZD_ * HD_);
    cvt(stream, (const float*)d_in[14], W_encs,   ZD_ * HD_);
    cvt(stream, (const float*)d_in[16], W_pri,    HD_ * HD_);
    cvt(stream, (const float*)d_in[18], W_prim,   ZD_ * HD_);
    cvt(stream, (const float*)d_in[20], W_pris,   ZD_ * HD_);
    cvt(stream, (const float*)d_in[22], W_dec1,   HD_ * 2 * HD_);
    cvt(stream, (const float*)d_in[24], W_dec2,   HD_ * HD_);
    cvt(stream, (const float*)d_in[26], W_decs,   XD_ * HD_);
    cvt(stream, (const float*)d_in[28], W_dm1,    XD_ * HD_);
    cvt(stream, (const float*)d_in[30], W_dm2,    XD_ * XD_);
    cvt(stream, (const float*)d_in[32], W_gih,    3 * HD_ * 2 * HD_);
    cvt(stream, (const float*)d_in[33], W_ghh,    3 * HD_ * HD_);
    cvt(stream, x, x_bf, T_ * B_ * XD_);
    hipLaunchKernelGGL(k_zero_h, dim3((B_ * HD_ + 255) / 256), dim3(256), 0, stream, h0, B_ * HD_);
    hipLaunchKernelGGL(k_zero_f, dim3(1), dim3(256), 0, stream, out, 3);

    for (int t = 0; t < T_; ++t) {
        const u16*   xb_t  = x_bf + (size_t)t * B_ * XD_;
        const float* x_t   = x    + (size_t)t * B_ * XD_;
        const float* eps_t = eps  + (size_t)t * B_ * ZD_;
        u16* h  = (t & 1) ? h1 : h0;
        u16* hn = (t & 1) ? h0 : h1;
        float* dmean_t = out + 3 + (size_t)t * B_ * XD_;

        // phi_x = relu(relu(x W1^T + b1) W2^T + b2)
        gemm(1, false, true, stream, xb_t, XD_, W_phi_x1, XD_, biasv[3], nullptr, tmp1, HD_, B_, HD_, XD_);
        gemm(1, false, true, stream, tmp1, HD_, W_phi_x2, HD_, biasv[5], nullptr, phi_x, HD_, B_, HD_, HD_);

        // enc: concat(phi_x, h) via column-split W1
        gemm(0, false, true, stream, phi_x, HD_, W_enc1,       2 * HD_, nullptr,  nullptr, enc_h1, HD_, B_, HD_, HD_);
        gemm(1, true,  true, stream, h,     HD_, W_enc1 + HD_, 2 * HD_, biasv[9], enc_h1,  enc_h1, HD_, B_, HD_, HD_);
        gemm(1, false, true, stream, enc_h1, HD_, W_enc2, HD_, biasv[11], nullptr, enc_h, HD_, B_, HD_, HD_);
        gemm(0, false, false, stream, enc_h, HD_, W_encm, HD_, biasv[13], nullptr, e_mean, ZD_, B_, ZD_, HD_);
        gemm(2, false, false, stream, enc_h, HD_, W_encs, HD_, biasv[15], nullptr, e_std,  ZD_, B_, ZD_, HD_);

        // prior
        gemm(1, false, true, stream, h, HD_, W_pri, HD_, biasv[17], nullptr, pri_h, HD_, B_, HD_, HD_);
        gemm(0, false, false, stream, pri_h, HD_, W_prim, HD_, biasv[19], nullptr, p_mean, ZD_, B_, ZD_, HD_);
        gemm(2, false, false, stream, pri_h, HD_, W_pris, HD_, biasv[21], nullptr, p_std,  ZD_, B_, ZD_, HD_);

        // z + phi_z
        hipLaunchKernelGGL(k_make_z, dim3((B_ * ZD_ + 255) / 256), dim3(256), 0, stream,
                           eps_t, e_mean, e_std, zbuf, B_ * ZD_);
        gemm(1, false, true, stream, zbuf, ZD_, W_phi_z, ZD_, biasv[7], nullptr, phi_z, HD_, B_, HD_, ZD_);

        // dec: concat(phi_z, h)
        gemm(0, false, true, stream, phi_z, HD_, W_dec1,       2 * HD_, nullptr,   nullptr, dec_h1, HD_, B_, HD_, HD_);
        gemm(1, true,  true, stream, h,     HD_, W_dec1 + HD_, 2 * HD_, biasv[23], dec_h1,  dec_h1, HD_, B_, HD_, HD_);
        gemm(1, false, true, stream, dec_h1, HD_, W_dec2, HD_, biasv[25], nullptr, dec_h, HD_, B_, HD_, HD_);
        gemm(1, false, true, stream, dec_h, HD_, W_dm1, HD_, biasv[29], nullptr, dm1, XD_, B_, XD_, HD_);
        gemm(0, false, false, stream, dm1, XD_, W_dm2, XD_, biasv[31], nullptr, dmean_t, XD_, B_, XD_, XD_);
        gemm(2, false, false, stream, dec_h, HD_, W_decs, HD_, biasv[27], nullptr, dstd, XD_, B_, XD_, HD_);

        // GRU: gx = [phi_x, phi_z] @ Wih^T (column-split), gh = h @ Whh^T
        gemm(0, false, false, stream, phi_x, HD_, W_gih,       2 * HD_, nullptr, nullptr, gx, 3 * HD_, B_, 3 * HD_, HD_);
        gemm(0, true,  false, stream, phi_z, HD_, W_gih + HD_, 2 * HD_, nullptr, gx,      gx, 3 * HD_, B_, 3 * HD_, HD_);
        gemm(0, false, false, stream, h, HD_, W_ghh, HD_, nullptr, nullptr, gh, 3 * HD_, B_, 3 * HD_, HD_);
        hipLaunchKernelGGL(k_gru, dim3((B_ * HD_ + 255) / 256), dim3(256), 0, stream,
                           gx, gh, h, hn, B_ * HD_);

        // losses
        hipLaunchKernelGGL(k_kld, dim3((B_ * ZD_ + 255) / 256), dim3(256), 0, stream,
                           e_mean, e_std, p_mean, p_std, out, B_ * ZD_);
        hipLaunchKernelGGL(k_nll_rec, dim3((B_ * XD_ + 255) / 256), dim3(256), 0, stream,
                           x_t, dmean_t, dstd, out, B_ * XD_);
    }
    (void)in_sizes; (void)n_in; (void)out_size; (void)ws_size;
}
